// Generator_15109694947587
// MI455X (gfx1250) — compile-verified, matrix operand order checked by reading, not run on hardware
//
#include <hip/hip_runtime.h>
#include <cmath>
#include <utility>

// ---------------------------------------------------------------------------
// Types
// ---------------------------------------------------------------------------
typedef __bf16 bf16;
typedef __attribute__((ext_vector_type(16))) __bf16 v16bf;
typedef __attribute__((ext_vector_type(8)))  float  v8f;
typedef __attribute__((ext_vector_type(4)))  unsigned int u32x4;

#define DEV static __device__ __forceinline__
DEV float b2f(bf16 x) { return (float)x; }
DEV bf16  f2b(float x) { return (bf16)x; }
DEV float lrelu(float x) { return x >= 0.f ? x : 0.2f * x; }

// ---------------------------------------------------------------------------
// Style MLP
// ---------------------------------------------------------------------------
__global__ void k_pixel_norm(const float* __restrict__ z, float* __restrict__ w) {
  __shared__ float red[256];
  int b = blockIdx.x;
  float s = 0.f;
  for (int j = threadIdx.x; j < 512; j += 256) { float v = z[b * 512 + j]; s += v * v; }
  red[threadIdx.x] = s;
  __syncthreads();
  for (int o = 128; o > 0; o >>= 1) {
    if (threadIdx.x < o) red[threadIdx.x] += red[threadIdx.x + o];
    __syncthreads();
  }
  float r = rsqrtf(red[0] / 512.f + 1e-8f);
  for (int j = threadIdx.x; j < 512; j += 256) w[b * 512 + j] = z[b * 512 + j] * r;
}

__global__ void k_mlp(const float* __restrict__ win, const float* __restrict__ Wm,
                      const float* __restrict__ bm, float* __restrict__ wout) {
  int idx = blockIdx.x * 256 + threadIdx.x;     // 16*512 exactly
  int b = idx >> 9, o = idx & 511;
  const float sl = 0.00044194173824159216f;      // (1/sqrt(512)) * lr_mlp
  float acc = 0.f;
  const float* wr = win + b * 512;
  const float* wm = Wm + (size_t)o * 512;
  for (int j = 0; j < 512; j++) acc = fmaf(wr[j], wm[j] * sl, acc);
  acc += bm[o] * 0.01f;
  wout[idx] = 1.414f * lrelu(acc);
}

// s[b,i] = sum_j wlat[b,j] * modW[i,j] / sqrt(512) + modb[i]
__global__ void k_style(const float* __restrict__ wlat, const float* __restrict__ modW,
                        const float* __restrict__ modb, float* __restrict__ s, int C) {
  int idx = blockIdx.x * 256 + threadIdx.x;
  if (idx >= 16 * C) return;
  int b = idx / C, i = idx % C;
  float acc = 0.f;
  const float* wr = wlat + b * 512;
  const float* mr = modW + (size_t)i * 512;
  for (int j = 0; j < 512; j++) acc = fmaf(wr[j], mr[j], acc);
  s[idx] = acc * 0.044194173824159216f + modb[i];
}

// ---------------------------------------------------------------------------
// Modulated-conv preparation
// ---------------------------------------------------------------------------
// q[oc,ic] = sum over 3x3 taps of weight^2
__global__ void k_qprep(const float* __restrict__ w, float* __restrict__ q, int total) {
  int idx = blockIdx.x * 256 + threadIdx.x;
  if (idx >= total) return;
  const float* p = w + (size_t)idx * 9;
  float acc = 0.f;
  for (int t = 0; t < 9; t++) acc = fmaf(p[t], p[t], acc);
  q[idx] = acc;
}

// d[b,oc] = rsqrt(sc2 * sum_ic q[oc,ic]*s[b,ic]^2 + 1e-8)
__global__ void k_demod(const float* __restrict__ q, const float* __restrict__ s,
                        float* __restrict__ d, int Cin, int M, float sc2) {
  int idx = blockIdx.x * 256 + threadIdx.x;
  if (idx >= 16 * M) return;
  int b = idx / M, oc = idx % M;
  const float* qp = q + (size_t)oc * Cin;
  const float* sp = s + b * Cin;
  float acc = 0.f;
  for (int i = 0; i < Cin; i++) { float sv = sp[i]; acc = fmaf(qp[i], sv * sv, acc); }
  d[idx] = rsqrtf(acc * sc2 + 1e-8f);
}

// Repack fp32 weights [oc,ic,3,3] -> bf16 A[oc, tap*Cin+ic] (tap-major K), optional flip.
__global__ void k_wprep(const float* __restrict__ w, bf16* __restrict__ A,
                        int Cin, int M, float sc, int flip) {
  int K = 9 * Cin;
  int idx = blockIdx.x * 256 + threadIdx.x;
  if (idx >= M * K) return;
  int oc = idx / K;
  int r = idx - oc * K;
  int tap = r / Cin;
  int ic = r - tap * Cin;
  int t = flip ? (8 - tap) : tap;
  A[idx] = f2b(w[((size_t)oc * Cin + ic) * 9 + t] * sc);
}

// const [512,4,4] (NCHW) -> padded NHWC bf16 act4 [16, 8, 8, 512] interior
__global__ void k_const_init(const float* __restrict__ cst, bf16* __restrict__ act) {
  int idx = blockIdx.x * 256 + threadIdx.x;   // 16*4*4*512
  if (idx >= 16 * 4 * 4 * 512) return;
  int c = idx & 511;
  int t = idx >> 9;
  int x = t & 3; t >>= 2;
  int y = t & 3;
  int b = t >> 2;
  act[(((size_t)b * 8 + y + 2) * 8 + (x + 2)) * 512 + c] = f2b(cst[c * 16 + y * 4 + x]);
}

// actMod = act * s[b,c] over the full padded extent (borders stay zero)
__global__ void k_modulate(const bf16* __restrict__ act, const float* __restrict__ s,
                           bf16* __restrict__ out, int HpWp, int C, int total) {
  int idx = blockIdx.x * 256 + threadIdx.x;
  if (idx >= total) return;
  int c = idx % C;
  int b = idx / (HpWp * C);
  out[idx] = f2b(b2f(act[idx]) * s[b * C + c]);
}

// Zero-dilation for conv_transpose: Dpad interior (2H-1)^2 at even positions = act*s
__global__ void k_dilate(const bf16* __restrict__ act, const float* __restrict__ s,
                         bf16* __restrict__ D, int H, int Hp, int Dp, int C, int total) {
  int idx = blockIdx.x * 256 + threadIdx.x;
  if (idx >= total) return;       // 16*H*H*C
  int c = idx % C;
  int t = idx / C;
  int x = t % H; t /= H;
  int y = t % H;
  int b = t / H;
  float v = b2f(act[(((size_t)b * Hp + y + 2) * Hp + (x + 2)) * C + c]) * s[b * C + c];
  D[(((size_t)b * Dp + 2 * y + 2) * Dp + (2 * x + 2)) * C + c] = f2b(v);
}

// ---------------------------------------------------------------------------
// Implicit-GEMM 3x3 conv with WMMA (bf16 in, fp32 accumulate)
//   out[n,m] = sum_k A[m,k] * patch(k,n),  K = 9*Cin (tap-major), N = 16*OH*OW
//   Block = 8 wave32s; wave tile = 64(M) x 16(N); block tile = 128(M) x 64(N).
//   K-loop unrolled x2 with double-buffered fragments so the loads of step
//   k+32 are in flight while the WMMAs of step k execute.  K-step indices
//   (tap ty/tx, ic0) are carried incrementally -- no div/mod in the hot loop.
//   A (weights, <=4.7MB) is L2-resident on MI455X (192MB L2), so no LDS
//   staging is needed; B (activations) streams from HBM at 23.3 TB/s.
// ---------------------------------------------------------------------------
__global__ __launch_bounds__(256)
void k_conv_gemm(const bf16* __restrict__ A, const bf16* __restrict__ act,
                 bf16* __restrict__ out,
                 int M, int Cin, int OH, int OW, int Hp, int Wp, int roff, int N) {
  const int K = 9 * Cin;                            // K % 64 == 0 (Cin in {256,512})
  const int wave = threadIdx.x >> 5;
  const int lane = threadIdx.x & 31;
  const int ln = lane & 15;
  const int kh = lane >> 4;                         // K-half select (ISA 7.12.2)
  const int mBase = blockIdx.x * 128 + (wave & 1) * 64;
  const int nBase = blockIdx.y * 64 + (wave >> 1) * 16;

  int n = nBase + ln;
  const bool nvalid = n < N;
  if (!nvalid) n = N - 1;                           // branchless clamp: EXEC stays full for WMMA
  const int ohw = OH * OW;
  const int b = n / ohw;
  const int r = n - b * ohw;
  const int oy = r / OW;
  const int ox = r - oy * OW;

  v8f acc0 = {0.f, 0.f, 0.f, 0.f, 0.f, 0.f, 0.f, 0.f};
  v8f acc1 = {0.f, 0.f, 0.f, 0.f, 0.f, 0.f, 0.f, 0.f};
  v8f acc2 = {0.f, 0.f, 0.f, 0.f, 0.f, 0.f, 0.f, 0.f};
  v8f acc3 = {0.f, 0.f, 0.f, 0.f, 0.f, 0.f, 0.f, 0.f};

  const bf16* Abase = A + (size_t)(mBase + ln) * K + kh * 8;
  const size_t aTile = (size_t)16 * K;
  const bf16* Bbase = act + ((((size_t)b * Hp + (oy + roff)) * Wp + (ox + roff)) * Cin) + kh * 8;
  const size_t rowStride = (size_t)Wp * Cin;

  union FragU { u32x4 u[2]; v16bf v; };
  FragU a0[4], a1[4], b0, b1;

  // incremental K-step state: tap = (ty,tx), channel block ic0, A offset kA
  int ty = 0, tx = 0, ic0 = 0;
  size_t kA = 0;

  auto loadFrags = [&](FragU* af, FragU& bfv) {
    const bf16* bp = Bbase + (size_t)ty * rowStride + (size_t)tx * Cin + ic0;
    bfv.u[0] = *(const u32x4*)bp;
    bfv.u[1] = *(const u32x4*)(bp + 16);
    const bf16* ap = Abase + kA;
#pragma unroll
    for (int t = 0; t < 4; t++) {
      af[t].u[0] = *(const u32x4*)ap;
      af[t].u[1] = *(const u32x4*)(ap + 16);
      ap += aTile;
    }
  };
  auto adv = [&]() {
    ic0 += 32;
    kA += 32;
    if (ic0 == Cin) {
      ic0 = 0;
      tx++;
      if (tx == 3) {
        tx = 0;
        ty++;
        if (ty == 3) { ty = 0; kA = 0; }   // wrap after last K-step (final preload is harmless)
      }
    }
  };

  loadFrags(a0, b0);
  adv();
  const int halves = K / 64;
  for (int it = 0; it < halves; it++) {
    // preload the odd half-step while computing the even one
    loadFrags(a1, b1);
    adv();
    acc0 = __builtin_amdgcn_wmma_f32_16x16x32_bf16(false, a0[0].v, false, b0.v, (short)0, acc0, false, false);
    acc1 = __builtin_amdgcn_wmma_f32_16x16x32_bf16(false, a0[1].v, false, b0.v, (short)0, acc1, false, false);
    acc2 = __builtin_amdgcn_wmma_f32_16x16x32_bf16(false, a0[2].v, false, b0.v, (short)0, acc2, false, false);
    acc3 = __builtin_amdgcn_wmma_f32_16x16x32_bf16(false, a0[3].v, false, b0.v, (short)0, acc3, false, false);
    // preload the next even half-step (wraps to step 0 on the last trip)
    loadFrags(a0, b0);
    adv();
    acc0 = __builtin_amdgcn_wmma_f32_16x16x32_bf16(false, a1[0].v, false, b1.v, (short)0, acc0, false, false);
    acc1 = __builtin_amdgcn_wmma_f32_16x16x32_bf16(false, a1[1].v, false, b1.v, (short)0, acc1, false, false);
    acc2 = __builtin_amdgcn_wmma_f32_16x16x32_bf16(false, a1[2].v, false, b1.v, (short)0, acc2, false, false);
    acc3 = __builtin_amdgcn_wmma_f32_16x16x32_bf16(false, a1[3].v, false, b1.v, (short)0, acc3, false, false);
  }

  if (nvalid) {
    // D layout: lane holds (m = 8*kh + r) within tile, n = ln
    bf16* op = out + (size_t)n * M + mBase + kh * 8;
    union { bf16 h[8]; u32x4 u; } st;
#pragma unroll
    for (int rr = 0; rr < 8; rr++) st.h[rr] = f2b(acc0[rr]);
    *(u32x4*)(op + 0) = st.u;
#pragma unroll
    for (int rr = 0; rr < 8; rr++) st.h[rr] = f2b(acc1[rr]);
    *(u32x4*)(op + 16) = st.u;
#pragma unroll
    for (int rr = 0; rr < 8; rr++) st.h[rr] = f2b(acc2[rr]);
    *(u32x4*)(op + 32) = st.u;
#pragma unroll
    for (int rr = 0; rr < 8; rr++) st.h[rr] = f2b(acc3[rr]);
    *(u32x4*)(op + 48) = st.u;
  }
}

// ---------------------------------------------------------------------------
// Epilogues
// ---------------------------------------------------------------------------
// Regular conv: demod + bias + leaky-relu*sqrt2, write padded NHWC bf16
__global__ void k_postconv(const bf16* __restrict__ tmp, const float* __restrict__ d,
                           const float* __restrict__ ab, bf16* __restrict__ act,
                           int H, int Hp, int M, int total) {
  int idx = blockIdx.x * 256 + threadIdx.x;
  if (idx >= total) return;            // 16*H*H*M
  int oc = idx % M;
  int n = idx / M;
  int x = n % H;
  int t = n / H;
  int y = t % H;
  int b = t / H;
  float v = b2f(tmp[(size_t)n * M + oc]) * d[b * M + oc] + ab[oc];
  act[(((size_t)b * Hp + y + 2) * Hp + (x + 2)) * M + oc] = f2b(1.41421356237f * lrelu(v));
}

// Up conv: 4x4 FIR blur over (2H+1)^2 pre-blur tensor, then demod + bias + lrelu*sqrt2
__global__ void k_postblur(const bf16* __restrict__ U, const float* __restrict__ d,
                           const float* __restrict__ ab, bf16* __restrict__ act,
                           int HO, int UH, int Hp, int M, int total) {
  int idx = blockIdx.x * 256 + threadIdx.x;
  if (idx >= total) return;            // 16*HO*HO*M
  int oc = idx % M;
  int n = idx / M;
  int x = n % HO;
  int t = n / HO;
  int y = t % HO;
  int b = t / HO;
  const float k4[4] = {0.25f, 0.75f, 0.75f, 0.25f};   // [1,3,3,1]/8 * 2  (K*4, separable)
  float acc = 0.f;
  for (int i = 0; i < 4; i++) {
    int yy = y + i - 1;
    if (yy < 0 || yy >= UH) continue;
    for (int j = 0; j < 4; j++) {
      int xx = x + j - 1;
      if (xx < 0 || xx >= UH) continue;
      acc += k4[i] * k4[j] * b2f(U[(((size_t)b * UH + yy) * UH + xx) * M + oc]);
    }
  }
  float v = acc * d[b * M + oc] + ab[oc];
  act[(((size_t)b * Hp + y + 2) * Hp + (x + 2)) * M + oc] = f2b(1.41421356237f * lrelu(v));
}

// to_rgb (1x1 modulated conv, no demod) + upsample-blur skip accumulate
__global__ void k_to_rgb(const bf16* __restrict__ act, const float* __restrict__ s,
                         const float* __restrict__ wrgb, const float* __restrict__ bias,
                         const float* __restrict__ skipPrev, float* __restrict__ out,
                         int H, int Hp, int C, int useSkip) {
  int idx = blockIdx.x * 256 + threadIdx.x;
  if (idx >= 16 * H * H) return;
  int x = idx % H;
  int t = idx / H;
  int y = t % H;
  int b = t / H;
  const bf16* ap = act + (((size_t)b * Hp + y + 2) * Hp + (x + 2)) * C;
  const float* sp = s + b * C;
  float a0 = 0.f, a1 = 0.f, a2 = 0.f;
  for (int c = 0; c < C; c++) {
    float v = b2f(ap[c]) * sp[c];
    a0 = fmaf(wrgb[c], v, a0);
    a1 = fmaf(wrgb[C + c], v, a1);
    a2 = fmaf(wrgb[2 * C + c], v, a2);
  }
  float sc = rsqrtf((float)C);
  float r0 = a0 * sc + bias[0];
  float r1 = a1 * sc + bias[1];
  float r2 = a2 * sc + bias[2];
  if (useSkip) {
    int Hh = H >> 1;
    const float k4[4] = {0.25f, 0.75f, 0.75f, 0.25f};
    for (int i = 0; i < 4; i++) {
      int yy = y + i - 2;
      if (yy < 0 || yy >= H || (yy & 1)) continue;
      for (int j = 0; j < 4; j++) {
        int xx = x + j - 2;
        if (xx < 0 || xx >= H || (xx & 1)) continue;
        float kk = k4[i] * k4[j];
        int yi = yy >> 1, xi = xx >> 1;
        r0 += kk * skipPrev[(((size_t)b * 3 + 0) * Hh + yi) * Hh + xi];
        r1 += kk * skipPrev[(((size_t)b * 3 + 1) * Hh + yi) * Hh + xi];
        r2 += kk * skipPrev[(((size_t)b * 3 + 2) * Hh + yi) * Hh + xi];
      }
    }
  }
  out[(((size_t)b * 3 + 0) * H + y) * H + x] = r0;
  out[(((size_t)b * 3 + 1) * H + y) * H + x] = r1;
  out[(((size_t)b * 3 + 2) * H + y) * H + x] = r2;
}

// ---------------------------------------------------------------------------
// Host orchestration
// Input flattening assumption: d_in[0]=z, then params with sorted dict keys:
//   const(1), conv1{act_b,mod_W,mod_b,noise_w,weight}(2..6),
//   convs[0..9] same 5-tuple (7+5i), mlp[0..7]{W,b} (57+2i),
//   to_rgb1{bias,mod_W,mod_b,weight}(73..76), to_rgbs[0..4] (77+4i)
// ---------------------------------------------------------------------------
extern "C" void kernel_launch(void* const* d_in, const int* in_sizes, int n_in,
                              void* d_out, int out_size, void* d_ws, size_t ws_size,
                              hipStream_t stream) {
  (void)in_sizes; (void)n_in; (void)out_size; (void)ws_size;
  const float* z   = (const float*)d_in[0];
  const float* cst = (const float*)d_in[1];

  // --- scratch arena ---
  char* cur = (char*)d_ws;
  auto alloc = [&](size_t bytes) -> void* {
    void* p = (void*)cur;
    cur += (bytes + 255) & ~(size_t)255;
    return p;
  };
  float* wbuf0 = (float*)alloc(16 * 512 * 4);
  float* wbuf1 = (float*)alloc(16 * 512 * 4);
  float* sbuf  = (float*)alloc(16 * 512 * 4);
  float* dbuf  = (float*)alloc(16 * 512 * 4);
  float* qbuf  = (float*)alloc((size_t)512 * 512 * 4);
  bf16*  Amat  = (bf16*)alloc((size_t)512 * 4608 * 2);
  bf16*  actMod = (bf16*)alloc((size_t)72000000 * 2);          // >= 16*132*132*256
  bf16*  Dpad  = (bf16*)alloc((size_t)141000000 * 2);          // >= 16*131*131*512
  bf16*  tmp   = (bf16*)alloc((size_t)69000000 * 2);           // >= 16*129*129*256
  size_t actElems[6] = {
      (size_t)16 * 8 * 8 * 512,     (size_t)16 * 12 * 12 * 512,
      (size_t)16 * 20 * 20 * 512,   (size_t)16 * 36 * 36 * 512,
      (size_t)16 * 68 * 68 * 512,   (size_t)16 * 132 * 132 * 256};
  bf16* acts[6];
  for (int i = 0; i < 6; i++) acts[i] = (bf16*)alloc(actElems[i] * 2);
  float* skipA = (float*)alloc((size_t)16 * 3 * 128 * 128 * 4);
  float* skipB = (float*)alloc((size_t)16 * 3 * 128 * 128 * 4);

  auto g1 = [](long long n) { return (unsigned)((n + 255) / 256); };

  // zero padded activation buffers (borders must be 0 every call; ws is poisoned)
  for (int i = 0; i < 6; i++) (void)hipMemsetAsync(acts[i], 0, actElems[i] * 2, stream);

  // --- style MLP ---
  k_pixel_norm<<<16, 256, 0, stream>>>(z, wbuf0);
  float* wa = wbuf0;
  float* wb = wbuf1;
  for (int l = 0; l < 8; l++) {
    k_mlp<<<32, 256, 0, stream>>>(wa, (const float*)d_in[57 + 2 * l],
                                  (const float*)d_in[58 + 2 * l], wb);
    std::swap(wa, wb);
  }
  float* wlat = wa;

  // --- layer helpers ---
  auto run_conv = [&](const float* const* P, const bf16* actIn, bf16* actOut,
                      int Cin, int Cout, int H, bool up) {
    const float* act_b = P[0];
    const float* modW = P[1];
    const float* modb = P[2];
    const float* wgt = P[4];
    int K = 9 * Cin;
    float sc = 1.0f / sqrtf((float)(Cin * 9));
    k_style<<<g1(16 * Cin), 256, 0, stream>>>(wlat, modW, modb, sbuf, Cin);
    k_qprep<<<g1((long long)Cout * Cin), 256, 0, stream>>>(wgt, qbuf, Cout * Cin);
    k_demod<<<g1(16 * Cout), 256, 0, stream>>>(qbuf, sbuf, dbuf, Cin, Cout, 1.0f / (Cin * 9.0f));
    k_wprep<<<g1((long long)Cout * K), 256, 0, stream>>>(wgt, Amat, Cin, Cout, sc, up ? 1 : 0);
    int Hp = H + 4;
    if (!up) {
      int total = 16 * Hp * Hp * Cin;
      k_modulate<<<g1(total), 256, 0, stream>>>(actIn, sbuf, actMod, Hp * Hp, Cin, total);
      int N = 16 * H * H;
      dim3 grid(Cout / 128, (N + 63) / 64);
      k_conv_gemm<<<grid, 256, 0, stream>>>(Amat, actMod, tmp, Cout, Cin, H, H, Hp, Hp, 1, N);
      int t2 = 16 * H * H * Cout;
      k_postconv<<<g1(t2), 256, 0, stream>>>(tmp, dbuf, act_b, actOut, H, Hp, Cout, t2);
    } else {
      int Dp = 2 * H + 3;
      (void)hipMemsetAsync(Dpad, 0, (size_t)16 * Dp * Dp * Cin * 2, stream);
      int total = 16 * H * H * Cin;
      k_dilate<<<g1(total), 256, 0, stream>>>(actIn, sbuf, Dpad, H, Hp, Dp, Cin, total);
      int UH = 2 * H + 1;
      int N = 16 * UH * UH;
      dim3 grid(Cout / 128, (N + 63) / 64);
      k_conv_gemm<<<grid, 256, 0, stream>>>(Amat, Dpad, tmp, Cout, Cin, UH, UH, Dp, Dp, 0, N);
      int HO = 2 * H, HpO = HO + 4;
      int t2 = 16 * HO * HO * Cout;
      k_postblur<<<g1(t2), 256, 0, stream>>>(tmp, dbuf, act_b, actOut, HO, UH, HpO, Cout, t2);
    }
  };
  auto run_rgb = [&](const float* const* P, const bf16* act, int C, int H,
                     const float* skipPrev, float* out) {
    k_style<<<g1(16 * C), 256, 0, stream>>>(wlat, P[1], P[2], sbuf, C);
    int total = 16 * H * H;
    k_to_rgb<<<g1(total), 256, 0, stream>>>(act, sbuf, P[3], P[0],
                                            skipPrev ? skipPrev : out, out,
                                            H, H + 4, C, skipPrev ? 1 : 0);
  };

  // --- generator body ---
  k_const_init<<<g1(16 * 4 * 4 * 512), 256, 0, stream>>>(cst, acts[0]);

  const float* conv1P[5];
  for (int j = 0; j < 5; j++) conv1P[j] = (const float*)d_in[2 + j];
  run_conv(conv1P, acts[0], acts[0], 512, 512, 4, false);

  const float* rgb1P[4];
  for (int j = 0; j < 4; j++) rgb1P[j] = (const float*)d_in[73 + j];
  run_rgb(rgb1P, acts[0], 512, 4, nullptr, skipA);

  int CH[6] = {512, 512, 512, 512, 512, 256};
  float* skips[2] = {skipB, skipA};
  const float* prevSkip = skipA;
  for (int i = 0; i < 5; i++) {
    int H = 4 << i;                 // input resolution
    int Cin = CH[i], Cout = CH[i + 1];
    const float* upP[5];
    for (int j = 0; j < 5; j++) upP[j] = (const float*)d_in[7 + 5 * (2 * i) + j];
    run_conv(upP, acts[i], acts[i + 1], Cin, Cout, H, true);
    const float* cP[5];
    for (int j = 0; j < 5; j++) cP[j] = (const float*)d_in[7 + 5 * (2 * i + 1) + j];
    run_conv(cP, acts[i + 1], acts[i + 1], Cout, Cout, 2 * H, false);
    const float* rP[4];
    for (int j = 0; j < 4; j++) rP[j] = (const float*)d_in[77 + 4 * i + j];
    float* outSkip = (i == 4) ? (float*)d_out : skips[i & 1];
    run_rgb(rP, acts[i + 1], Cout, 2 * H, prevSkip, outSkip);
    prevSkip = outSkip;
  }
}
